// LlamaAttention_1821066133586
// MI455X (gfx1250) — compile-verified, matrix operand order checked by reading, not run on hardware
//
#include <hip/hip_runtime.h>
#include <hip/hip_bf16.h>

// ---------------------------------------------------------------------------
// LlamaAttention forward for MI455X (gfx1250, wave32, WMMA + TDM).
// - All GEMMs (QKV proj, scores, PV, O proj) on v_wmma_f32_16x16x32_bf16,
//   f32 accumulation (bf16 inputs converted once from the fp32 operands).
// - Projection GEMMs: Tensor Data Mover streams 128x32 X/W tiles into
//   double-buffered LDS (TENSORcnt + workgroup barrier), waves consume with
//   ds_load_b128 -> WMMA.
// - Attention: flash-style online softmax, scores computed transposed so the
//   key-dim reduction is 8 regs + one shfl_xor(16); causal mask only touches
//   the single diagonal chunk (main key loop is mask-free).
// ---------------------------------------------------------------------------

typedef __bf16 v16bf __attribute__((ext_vector_type(16)));
typedef float  v8f   __attribute__((ext_vector_type(8)));
typedef unsigned int u32x4 __attribute__((ext_vector_type(4)));
typedef unsigned int u32x2 __attribute__((ext_vector_type(2)));
typedef int          i32x4 __attribute__((ext_vector_type(4)));
typedef int          i32x8 __attribute__((ext_vector_type(8)));

#define BATCH  2
#define SEQ    2048
#define DMODEL 2048
#define NHEAD  16
#define HEADD  128
#define MROWS  (BATCH * SEQ)   // 4096

__device__ __forceinline__ unsigned short f2bf(float f) {
  unsigned int u = __float_as_uint(f);
  u += 0x7FFFu + ((u >> 16) & 1u);          // round-to-nearest-even
  return (unsigned short)(u >> 16);
}
__device__ __forceinline__ float bf2f(unsigned short h) {
  return __uint_as_float(((unsigned int)h) << 16);
}

union Frag {
  v16bf v;
  u32x4 q[2];
  unsigned short u[16];
};

// A-style 16x32 bf16 fragment loader (row-major tile, leading dim `ld` elems).
// lane m = lane&15 is the row; halves 0-7 hold K=kb..kb+7, halves 8-15 hold
// K=16+kb..23+kb, kb = (lane>=16)*8. Two b128 loads (global or ds).
template <typename PT>
__device__ __forceinline__ v16bf load_frag(const PT* base, int ld) {
  const int lane = threadIdx.x & 31;
  const int m  = lane & 15;
  const int kb = (lane >> 4) << 3;
  Frag f;
  f.q[0] = *reinterpret_cast<const u32x4*>(base + (size_t)m * ld + kb);
  f.q[1] = *reinterpret_cast<const u32x4*>(base + (size_t)m * ld + 16 + kb);
  return f.v;
}

// ---------------------------------------------------------------------------
// Tensor Data Mover: 2D tile (tile_w x tile_h elements of 2 bytes) from a
// row-major global tensor (row stride = stride_elems) into LDS at lds_off.
// D# bit packing per CDNA5 ISA 8.3/8.4 (group0 128b, group1 256b).
// ---------------------------------------------------------------------------
__device__ __forceinline__ void tdm_load_2d_bf16(
    unsigned lds_off, const void* gaddr, unsigned tile_w, unsigned tile_h,
    unsigned stride_elems, unsigned tensor_w, unsigned tensor_h) {
  const unsigned long long ga = (unsigned long long)(uintptr_t)gaddr;
  u32x4 g0;
  g0[0] = 1u;                                        // count=1 (user D#)
  g0[1] = lds_off;                                   // lds_addr (bytes)
  g0[2] = (unsigned)(ga & 0xFFFFFFFFu);              // global_addr[31:0]
  g0[3] = (unsigned)((ga >> 32) & 0x1FFFFFFu)        // global_addr[56:32]
          | (2u << 30);                              // type=2 ("image")
  i32x8 g1;
  g1[0] = (int)(1u << 16);                           // data_size=1 (2B)
  g1[1] = (int)((tensor_w & 0xFFFFu) << 16);         // tensor_dim0[15:0]
  g1[2] = (int)((tensor_w >> 16) | ((tensor_h & 0xFFFFu) << 16));
  g1[3] = (int)((tensor_h >> 16) | (tile_w << 16));  // tile_dim0
  g1[4] = (int)(tile_h);                             // tile_dim1 (tile_dim2=0)
  g1[5] = (int)(stride_elems);                       // tensor_dim0_stride[31:0]
  g1[6] = 0;
  g1[7] = 0;
  i32x4 z4 = {0, 0, 0, 0};
#if defined(__clang_major__) && (__clang_major__ >= 23)
  i32x8 z8 = {0, 0, 0, 0, 0, 0, 0, 0};
  __builtin_amdgcn_tensor_load_to_lds(g0, g1, z4, z4, z8, 0);
#else
  __builtin_amdgcn_tensor_load_to_lds(g0, g1, z4, z4, 0);
#endif
}

// ---------------------------------------------------------------------------
// fp32 -> bf16 conversion, 4 elements / thread (b128 in, b64 out)
// ---------------------------------------------------------------------------
__global__ __launch_bounds__(256) void cvt_f32_bf16_x4(
    const float* __restrict__ in, unsigned short* __restrict__ out, int n4) {
  int i = blockIdx.x * blockDim.x + threadIdx.x;
  if (i >= n4) return;
  const float4 v = reinterpret_cast<const float4*>(in)[i];
  u32x2 p;
  p[0] = (unsigned)f2bf(v.x) | ((unsigned)f2bf(v.y) << 16);
  p[1] = (unsigned)f2bf(v.z) | ((unsigned)f2bf(v.w) << 16);
  reinterpret_cast<u32x2*>(out)[i] = p;
}

// ---------------------------------------------------------------------------
// WMMA GEMM with TDM->LDS double buffering:
//   Out[m,n] = sum_d X[m,d] * W[n,d]   (einsum 'md,nd->mn')
// Block = 256 thr (8 waves), block tile 128x128, wave tile 64x32 (4x2 WMMA).
// Per k-step: wave0 issues 2 TDM descriptors for the next 128x32 X/W tiles;
// all waves compute from the current LDS buffers (ds_load_b128 fragments).
// Dynamic LDS: 4 buffers of 8 KB (Xcur, Xnxt, Wcur, Wnxt) = 32 KB.
// ---------------------------------------------------------------------------
#define GTILE_BYTES (128 * 32 * 2)   // 8 KB

template <bool OUT_F32>
__global__ __launch_bounds__(256) void gemm_wmma(
    const unsigned short* __restrict__ X,
    const unsigned short* __restrict__ W,
    void* __restrict__ Out, int M, int N, int K) {
  extern __shared__ unsigned short smem[];   // 32 KB dynamic
  const int lane = threadIdx.x & 31;
  const int wid  = threadIdx.x >> 5;   // 0..7
  const int wr   = wid >> 2;           // 0..1  (64 rows each)
  const int wc   = wid & 3;            // 0..3  (32 cols each)
  const int Mblk = blockIdx.y * 128;
  const int Nblk = blockIdx.x * 128;
  const int M0 = Mblk + wr * 64;
  const int N0 = Nblk + wc * 32;

  // LDS byte offsets of the 4 tile buffers (dynamic LDS starts at offset 0).
  const unsigned XOFF[2] = {0u, (unsigned)GTILE_BYTES};
  const unsigned WOFF[2] = {(unsigned)(2 * GTILE_BYTES), (unsigned)(3 * GTILE_BYTES)};

  v8f c[4][2];
#pragma unroll
  for (int i = 0; i < 4; i++)
#pragma unroll
    for (int j = 0; j < 2; j++)
#pragma unroll
      for (int r = 0; r < 8; r++) c[i][j][r] = 0.f;

  const int NK = K / 32;
  if (wid == 0) {
    tdm_load_2d_bf16(XOFF[0], X + (size_t)Mblk * K, 32, 128, K, K, M);
    tdm_load_2d_bf16(WOFF[0], W + (size_t)Nblk * K, 32, 128, K, K, N);
    __builtin_amdgcn_s_wait_tensorcnt(0);
  }
  __syncthreads();

  for (int it = 0; it < NK; it++) {
    const int cur = it & 1;
    if (wid == 0 && it + 1 < NK) {
      const int k1 = (it + 1) * 32;
      tdm_load_2d_bf16(XOFF[cur ^ 1], X + (size_t)Mblk * K + k1, 32, 128, K, K, M);
      tdm_load_2d_bf16(WOFF[cur ^ 1], W + (size_t)Nblk * K + k1, 32, 128, K, K, N);
    }

    const unsigned short* Xs = smem + (XOFF[cur] >> 1);
    const unsigned short* Ws = smem + (WOFF[cur] >> 1);
    v16bf a[4], bfr[2];
#pragma unroll
    for (int i = 0; i < 4; i++)
      a[i] = load_frag(Xs + (size_t)(wr * 64 + 16 * i) * 32, 32);
#pragma unroll
    for (int j = 0; j < 2; j++)
      bfr[j] = load_frag(Ws + (size_t)(wc * 32 + 16 * j) * 32, 32);
#pragma unroll
    for (int i = 0; i < 4; i++)
#pragma unroll
      for (int j = 0; j < 2; j++)
        c[i][j] = __builtin_amdgcn_wmma_f32_16x16x32_bf16(
            false, a[i], false, bfr[j], (short)0, c[i][j], false, false);

    if (wid == 0 && it + 1 < NK) __builtin_amdgcn_s_wait_tensorcnt(0);
    __syncthreads();
  }

  const int hi = (lane >> 4) & 1;
  const int cl = lane & 15;
#pragma unroll
  for (int i = 0; i < 4; i++)
#pragma unroll
    for (int j = 0; j < 2; j++)
#pragma unroll
      for (int r = 0; r < 8; r++) {
        const int row = M0 + 16 * i + r + 8 * hi;
        const int col = N0 + 16 * j + cl;
        if (OUT_F32)
          ((float*)Out)[(size_t)row * N + col] = c[i][j][r];
        else
          ((unsigned short*)Out)[(size_t)row * N + col] = f2bf(c[i][j][r]);
      }
}

// ---------------------------------------------------------------------------
// RoPE + rearrange: qraw/kraw [B*S, D] -> Qh/Kh [B,H,S,HD] (rotated; Q is
// pre-scaled by 1/sqrt(HD)), vraw -> VT [B,H,HD,S] (transposed).
// One thread per (b,s,h,i), i in [0,64).
// ---------------------------------------------------------------------------
__global__ __launch_bounds__(256) void rope_rearrange(
    const unsigned short* __restrict__ Qr, const unsigned short* __restrict__ Kr,
    const unsigned short* __restrict__ Vr, const int* __restrict__ pos_ids,
    unsigned short* __restrict__ Qh, unsigned short* __restrict__ Kh,
    unsigned short* __restrict__ VT) {
  const int idx = blockIdx.x * blockDim.x + threadIdx.x;   // B*S*H*64 threads
  const int i = idx & 63;
  const int h = (idx >> 6) & (NHEAD - 1);
  const int s = (idx >> 10) & (SEQ - 1);
  const int b = idx >> 21;

  const float pos = (float)pos_ids[b * SEQ + s];
  // inv_freq = theta^(-2i/128) = exp(-(2i/128)*ln(10000))
  const float invf = __expf(-(float)(2 * i) * (1.0f / 128.0f) * 9.210340371976184f);
  float sn, cs;
  __sincosf(pos * invf, &sn, &cs);
  const float qscale = 0.08838834764831845f;   // 1/sqrt(128), folded into Q

  const size_t src = ((size_t)(b * SEQ + s)) * DMODEL + h * HEADD;
  const size_t dst = (((size_t)(b * NHEAD + h)) * SEQ + s) * HEADD;

  const float q1 = bf2f(Qr[src + i]), q2 = bf2f(Qr[src + 64 + i]);
  Qh[dst + i]      = f2bf((q1 * cs - q2 * sn) * qscale);
  Qh[dst + 64 + i] = f2bf((q2 * cs + q1 * sn) * qscale);

  const float k1 = bf2f(Kr[src + i]), k2 = bf2f(Kr[src + 64 + i]);
  Kh[dst + i]      = f2bf(k1 * cs - k2 * sn);
  Kh[dst + 64 + i] = f2bf(k2 * cs + k1 * sn);

  const size_t vtb = ((size_t)(b * NHEAD + h)) * HEADD;
  VT[(vtb + i) * SEQ + s]      = Vr[src + i];
  VT[(vtb + 64 + i) * SEQ + s] = Vr[src + 64 + i];
}

// ---------------------------------------------------------------------------
// Flash attention, per wave: one 16-query tile vs all causal keys.
// S^T[k,q] = K_tile x Q_tile^T (Q pre-scaled). lane = query column; key index
// = reg r + 8*(lane>=16); softmax over keys = 8-reg reduce + shfl_xor(16).
// The two 16x16 S^T tiles map 1:1 onto one K=32 B fragment (halves 0-7/8-15).
// out^T(128x16) += V^T x P^T, 8 WMMA / 32-key chunk.
// Causal mask applied only to the single diagonal (last) chunk.
// ---------------------------------------------------------------------------
__global__ __launch_bounds__(128) void attn_wmma(
    const unsigned short* __restrict__ Qh,  // [B,H,S,HD]
    const unsigned short* __restrict__ Kh,  // [B,H,S,HD]
    const unsigned short* __restrict__ VT,  // [B,H,HD,S]
    unsigned short* __restrict__ AO) {      // [B,S,H*HD]
  const int lane = threadIdx.x & 31;
  const int wid  = threadIdx.x >> 5;       // 4 waves, 4 query tiles per block
  const int bh = blockIdx.x;               // b*H + h
  const int b = bh / NHEAD, h = bh % NHEAD;
  const int q0 = (blockIdx.y * 4 + wid) * 16;
  const int hi = (lane >> 4) & 1;
  const int ql = lane & 15;                // this lane's query column

  const size_t qbase  = ((size_t)bh * SEQ + q0) * HEADD;
  const size_t kbase  = (size_t)bh * SEQ * HEADD;
  const size_t vtbase = (size_t)bh * HEADD * SEQ;

  v16bf qf[4];
#pragma unroll
  for (int dc = 0; dc < 4; dc++)
    qf[dc] = load_frag(Qh + qbase + dc * 32, HEADD);

  v8f o[8];
#pragma unroll
  for (int t = 0; t < 8; t++)
#pragma unroll
    for (int r = 0; r < 8; r++) o[t][r] = 0.f;
  float m_run = -1e30f, l_run = 0.f;

  const int kv_len  = q0 + 16;                  // causal: keys <= query
  const int last_kc = (((kv_len + 31) >> 5) - 1) << 5;

  // ---- main loop: chunks entirely below the diagonal (no masking) ----
  for (int kc = 0; kc < last_kc; kc += 32) {
    v8f s0, s1;
#pragma unroll
    for (int r = 0; r < 8; r++) { s0[r] = 0.f; s1[r] = 0.f; }

    const unsigned short* kp = Kh + kbase + (size_t)kc * HEADD;
    __builtin_prefetch(kp + 32 * HEADD + (size_t)lane * HEADD, 0, 0);
#pragma unroll
    for (int dc = 0; dc < 4; dc++) {
      v16bf a0 = load_frag(kp + dc * 32, HEADD);
      v16bf a1 = load_frag(kp + 16 * HEADD + dc * 32, HEADD);
      s0 = __builtin_amdgcn_wmma_f32_16x16x32_bf16(
          false, a0, false, qf[dc], (short)0, s0, false, false);
      s1 = __builtin_amdgcn_wmma_f32_16x16x32_bf16(
          false, a1, false, qf[dc], (short)0, s1, false, false);
    }

    float tmax = -1e30f;
#pragma unroll
    for (int r = 0; r < 8; r++) tmax = fmaxf(tmax, fmaxf(s0[r], s1[r]));
    tmax = fmaxf(tmax, __shfl_xor(tmax, 16, 32));
    const float m_new = fmaxf(m_run, tmax);
    const float corr  = __expf(m_run - m_new);
#pragma unroll
    for (int t = 0; t < 8; t++)
#pragma unroll
      for (int r = 0; r < 8; r++) o[t][r] *= corr;
    l_run *= corr;

    Frag pf;
    float lsum = 0.f;
#pragma unroll
    for (int r = 0; r < 8; r++) {
      const float p0 = __expf(s0[r] - m_new);
      const float p1 = __expf(s1[r] - m_new);
      lsum += p0 + p1;
      pf.u[r]     = f2bf(p0);
      pf.u[8 + r] = f2bf(p1);
    }
    lsum += __shfl_xor(lsum, 16, 32);
    l_run += lsum;
    m_run = m_new;

#pragma unroll
    for (int t = 0; t < 8; t++) {
      v16bf vf = load_frag(VT + vtbase + (size_t)(t * 16) * SEQ + kc, SEQ);
      o[t] = __builtin_amdgcn_wmma_f32_16x16x32_bf16(
          false, vf, false, pf.v, (short)0, o[t], false, false);
    }
  }

  // ---- tail chunk: crosses the diagonal, apply causal mask ----
  {
    const int kc = last_kc;
    v8f s0, s1;
#pragma unroll
    for (int r = 0; r < 8; r++) { s0[r] = 0.f; s1[r] = 0.f; }

    const unsigned short* kp = Kh + kbase + (size_t)kc * HEADD;
#pragma unroll
    for (int dc = 0; dc < 4; dc++) {
      v16bf a0 = load_frag(kp + dc * 32, HEADD);
      v16bf a1 = load_frag(kp + 16 * HEADD + dc * 32, HEADD);
      s0 = __builtin_amdgcn_wmma_f32_16x16x32_bf16(
          false, a0, false, qf[dc], (short)0, s0, false, false);
      s1 = __builtin_amdgcn_wmma_f32_16x16x32_bf16(
          false, a1, false, qf[dc], (short)0, s1, false, false);
    }

    const int qg = q0 + ql;
    float tmax = -1e30f;
#pragma unroll
    for (int r = 0; r < 8; r++) {
      const int k0g = kc + r + 8 * hi;
      const float v0 = s0[r] + ((k0g      > qg) ? -1e9f : 0.f);
      const float v1 = s1[r] + ((k0g + 16 > qg) ? -1e9f : 0.f);
      s0[r] = v0; s1[r] = v1;
      tmax = fmaxf(tmax, fmaxf(v0, v1));
    }
    tmax = fmaxf(tmax, __shfl_xor(tmax, 16, 32));
    const float m_new = fmaxf(m_run, tmax);
    const float corr  = __expf(m_run - m_new);
#pragma unroll
    for (int t = 0; t < 8; t++)
#pragma unroll
      for (int r = 0; r < 8; r++) o[t][r] *= corr;
    l_run *= corr;

    Frag pf;
    float lsum = 0.f;
#pragma unroll
    for (int r = 0; r < 8; r++) {
      const float p0 = __expf(s0[r] - m_new);
      const float p1 = __expf(s1[r] - m_new);
      lsum += p0 + p1;
      pf.u[r]     = f2bf(p0);
      pf.u[8 + r] = f2bf(p1);
    }
    lsum += __shfl_xor(lsum, 16, 32);
    l_run += lsum;

#pragma unroll
    for (int t = 0; t < 8; t++) {
      v16bf vf = load_frag(VT + vtbase + (size_t)(t * 16) * SEQ + kc, SEQ);
      o[t] = __builtin_amdgcn_wmma_f32_16x16x32_bf16(
          false, vf, false, pf.v, (short)0, o[t], false, false);
    }
  }

  // epilogue: normalize and write AO[b, q0+ql, h*128 + d] (8 bf16 = b128/lane)
  const float inv = 1.f / l_run;
  const size_t orow = ((size_t)b * SEQ + (q0 + ql)) * DMODEL + h * HEADD;
#pragma unroll
  for (int t = 0; t < 8; t++) {
    Frag w;
#pragma unroll
    for (int r = 0; r < 8; r++) w.u[r] = f2bf(o[t][r] * inv);
    *reinterpret_cast<u32x4*>(&AO[orow + t * 16 + 8 * hi]) = w.q[0];
  }
}

// ---------------------------------------------------------------------------
// Host-side orchestration
// ---------------------------------------------------------------------------
extern "C" void kernel_launch(void* const* d_in, const int* in_sizes, int n_in,
                              void* d_out, int out_size, void* d_ws, size_t ws_size,
                              hipStream_t stream) {
  const float* hs  = (const float*)d_in[0];
  // d_in[1] = attention_mask: causal, applied analytically in attn_wmma.
  const int*   pos = (const int*)d_in[2];
  const float* Wq  = (const float*)d_in[3];
  const float* Wk  = (const float*)d_in[4];
  const float* Wv  = (const float*)d_in[5];
  const float* Wo  = (const float*)d_in[6];

  char* ws = (char*)d_ws;
  size_t off = 0;
  auto alloc = [&](size_t bytes) -> void* {
    void* p = ws + off;
    off += (bytes + 255) & ~(size_t)255;
    return p;
  };
  const size_t ACT = (size_t)MROWS * DMODEL * sizeof(unsigned short);   // 16 MB
  const size_t WSZ = (size_t)DMODEL * DMODEL * sizeof(unsigned short);  //  8 MB
  unsigned short* hsb  = (unsigned short*)alloc(ACT);
  unsigned short* wqb  = (unsigned short*)alloc(WSZ);
  unsigned short* wkb  = (unsigned short*)alloc(WSZ);
  unsigned short* wvb  = (unsigned short*)alloc(WSZ);
  unsigned short* wob  = (unsigned short*)alloc(WSZ);
  unsigned short* qraw = (unsigned short*)alloc(ACT);
  unsigned short* kraw = (unsigned short*)alloc(ACT);
  unsigned short* vraw = (unsigned short*)alloc(ACT);
  unsigned short* qh   = (unsigned short*)alloc(ACT);
  unsigned short* kh   = (unsigned short*)alloc(ACT);
  unsigned short* vt   = (unsigned short*)alloc(ACT);
  unsigned short* ao   = (unsigned short*)alloc(ACT);
  (void)ws_size; (void)in_sizes; (void)n_in; (void)out_size;

  // 1) fp32 -> bf16
  const int nAct4 = (MROWS * DMODEL) / 4;
  const int nW4   = (DMODEL * DMODEL) / 4;
  cvt_f32_bf16_x4<<<(nAct4 + 255) / 256, 256, 0, stream>>>(hs, hsb, nAct4);
  cvt_f32_bf16_x4<<<(nW4 + 255) / 256, 256, 0, stream>>>(Wq, wqb, nW4);
  cvt_f32_bf16_x4<<<(nW4 + 255) / 256, 256, 0, stream>>>(Wk, wkb, nW4);
  cvt_f32_bf16_x4<<<(nW4 + 255) / 256, 256, 0, stream>>>(Wv, wvb, nW4);
  cvt_f32_bf16_x4<<<(nW4 + 255) / 256, 256, 0, stream>>>(Wo, wob, nW4);

  // 2) Q/K/V projections (TDM->LDS double-buffered WMMA GEMM, bf16 out)
  dim3 gg(DMODEL / 128, MROWS / 128);  // (16, 32)
  const size_t lds_bytes = 4 * GTILE_BYTES;  // 32 KB
  gemm_wmma<false><<<gg, 256, lds_bytes, stream>>>(hsb, wqb, qraw, MROWS, DMODEL, DMODEL);
  gemm_wmma<false><<<gg, 256, lds_bytes, stream>>>(hsb, wkb, kraw, MROWS, DMODEL, DMODEL);
  gemm_wmma<false><<<gg, 256, lds_bytes, stream>>>(hsb, wvb, vraw, MROWS, DMODEL, DMODEL);

  // 3) RoPE + head-major rearrange (+ V transpose, Q pre-scaled)
  rope_rearrange<<<(BATCH * SEQ * NHEAD * 64) / 256, 256, 0, stream>>>(
      qraw, kraw, vraw, pos, qh, kh, vt);

  // 4) Flash attention (WMMA)
  dim3 ga(BATCH * NHEAD, SEQ / 64);  // (32, 32), 4 waves/block, 16 queries/wave
  attn_wmma<<<ga, 128, 0, stream>>>(qh, kh, vt, ao);

  // 5) Output projection (WMMA GEMM, fp32 out)
  gemm_wmma<true><<<gg, 256, lds_bytes, stream>>>(ao, wob, d_out, MROWS, DMODEL, DMODEL);
}
